// NodeMemory_75265006895940
// MI455X (gfx1250) — matrix operand order repaired and checked by reading, not compiled, not run on hardware
//
#include <hip/hip_runtime.h>

// ---------------- problem constants (match reference) ----------------
constexpr int cN = 10000;
constexpr int cE = 100000;
constexpr int cD = 128;
constexpr int cL = 4;
constexpr int cF = 128;
constexpr int cR = 100;
constexpr int cH = 8;
constexpr int RAW  = cL * cD + cF;   // 640
constexpr int RAW1 = RAW + 1;        // 641
constexpr float NEGV  = -1e30f;
constexpr float SLOPE = 0.01f;       // jax.nn.leaky_relu default

__constant__ float c_lambs[4] = {1.0f, 0.1f, 0.01f, 0.001f};

typedef __attribute__((ext_vector_type(16))) _Float16 v16h;
typedef __attribute__((ext_vector_type(8)))  _Float16 v8h;
typedef __attribute__((ext_vector_type(8)))  float    v8f;
union V16H { v16h v; v8h p[2]; _Float16 h[16]; };
union V8F  { v8f  v; float    f[8];  };

static inline int divup(long a, long b) { return (int)((a + b - 1) / b); }

// ---------------- generic elementwise kernels ----------------
__global__ void k_zero(float* p, size_t n) {
  size_t i = blockIdx.x * (size_t)blockDim.x + threadIdx.x;
  if (i < n) p[i] = 0.0f;
}

__global__ void k_cvt_f16(const float* __restrict__ s, _Float16* __restrict__ d, size_t n) {
  size_t i = blockIdx.x * (size_t)blockDim.x + threadIdx.x;
  if (i < n) d[i] = (_Float16)s[i];
}

// dst[n][k] = (k < K) ? src[k][n] : 0    (transpose + zero-pad K -> Kpad)
__global__ void k_cvt_T_pad(const float* __restrict__ src, _Float16* __restrict__ dst,
                            int K, int Nc, int Kpad) {
  size_t i = blockIdx.x * (size_t)blockDim.x + threadIdx.x;
  size_t tot = (size_t)Nc * Kpad;
  if (i >= tot) return;
  int n = (int)(i / Kpad), k = (int)(i % Kpad);
  dst[i] = (k < K) ? (_Float16)src[(size_t)k * Nc + n] : (_Float16)0.0f;
}

// ---------------- stage 1: segment max / count ----------------
__global__ void k_seg(const float* __restrict__ ts, const int* __restrict__ src,
                      float* t_u, float* cnt, int E) {
  int e = blockIdx.x * blockDim.x + threadIdx.x;
  if (e >= E) return;
  int s = src[e];
  atomicMax((int*)&t_u[s], __float_as_int(ts[e]));  // ts >= 0, bits order-preserving
  atomicAdd(&cnt[s], 1.0f);
}

__global__ void k_weights(const float* __restrict__ ts, const float* __restrict__ etd,
                          const int* __restrict__ src, const float* __restrict__ t_u,
                          float* __restrict__ wdec, float* __restrict__ wts, int E) {
  int e = blockIdx.x * blockDim.x + threadIdx.x;
  if (e >= E) return;
  float tu = t_u[src[e]], d = etd[e], dt = tu - ts[e];
#pragma unroll
  for (int l = 0; l < 4; ++l) {
    wdec[e * 4 + l] = __expf(-c_lambs[l] * d);
    wts[e * 4 + l]  = __expf(-c_lambs[l] * dt);
  }
}

// fused raw construction + 4x weighted segment scatter-add
__global__ __launch_bounds__(128)
void k_scatter(const float* __restrict__ memory, const float* __restrict__ feat,
               const int* __restrict__ src, const int* __restrict__ tgt,
               const float* __restrict__ wdec, const float* __restrict__ wts,
               float* __restrict__ agg) {
  int e = blockIdx.x, chunk = blockIdx.y, t = threadIdx.x;
  int s = src[e], g = tgt[e];
  float val; int c;
  if (chunk < 4) {                      // memory[target] * exp(-lamb*etd)
    c = chunk * 128 + t;
    val = memory[(size_t)g * 512 + c] * wdec[e * 4 + chunk];
  } else if (chunk == 4) {              // node_features[target]
    c = 512 + t;
    val = feat[(size_t)g * 128 + t];
  } else {                              // ones column
    if (t != 0) return;
    c = 640; val = 1.0f;
  }
  float w0 = wts[e * 4 + 0], w1 = wts[e * 4 + 1], w2 = wts[e * 4 + 2], w3 = wts[e * 4 + 3];
  float* base = agg + (size_t)s * (4 * RAW1) + c;
  atomicAdd(base + 0 * RAW1, val * w0);
  atomicAdd(base + 1 * RAW1, val * w1);
  atomicAdd(base + 2 * RAW1, val * w2);
  atomicAdd(base + 3 * RAW1, val * w3);
}

// agg_m (stride 641 -> dense 640) to f16; optionally normalized by agg_c
__global__ void k_aggm_f16(const float* __restrict__ agg, _Float16* __restrict__ dst,
                           int rows, int normalize) {
  size_t i = blockIdx.x * (size_t)blockDim.x + threadIdx.x;
  size_t tot = (size_t)rows * RAW;
  if (i >= tot) return;
  int r = (int)(i / RAW), c = (int)(i % RAW);
  float v = agg[(size_t)r * RAW1 + c];
  if (normalize) {
    float den = agg[(size_t)r * RAW1 + RAW];
    v /= (den > 0.0f) ? den : 1.0f;
  }
  dst[i] = (_Float16)v;
}

// ---------------- WMMA f16 GEMM:  C = act(A @ B + bias) ----------------
// A: M x K f16 row-major.  BT: Nc x K f16 (B transposed).  C row stride = ldc.
// One wave computes a 16x32 output tile (two accumulators share the A fragment).
// Out-of-range A rows are clamped to row 0: they only affect out-of-range D rows,
// which are masked at the store, so no zero-fill (and no divergence) is needed.
// act: 0=none 1=relu 2=leaky_relu(0.01)
__global__ __launch_bounds__(128)
void k_wmma_gemm(const _Float16* __restrict__ A, const _Float16* __restrict__ BT,
                 const float* __restrict__ bias, float* __restrict__ C,
                 int M, int K, int Nc, int ldc, int act) {
  const int lane = threadIdx.x;                 // 0..31, one wave per 16x32 tile
  const int tilesN2 = Nc >> 5;                  // Nc is always a multiple of 32
  const int tilesM  = (M + 15) >> 4;
  const int tileId = blockIdx.x * 4 + threadIdx.y;
  if (tileId >= tilesM * tilesN2) return;
  const int tm = tileId / tilesN2, tn = tileId % tilesN2;

  const int half = (lane >> 4) & 1;             // lane group 0-15 / 16-31
  const int l15  = lane & 15;
  const int arow = tm * 16 + l15;
  const _Float16* aptr  = A  + (size_t)(arow < M ? arow : 0) * K;
  const _Float16* bptr0 = BT + (size_t)(tn * 32 + l15) * K;
  const _Float16* bptr1 = bptr0 + (size_t)16 * K;

  V8F acc0, acc1;
#pragma unroll
  for (int i = 0; i < 8; ++i) { acc0.f[i] = 0.0f; acc1.f[i] = 0.0f; }
  v8f c0 = acc0.v, c1 = acc1.v;

  for (int k0 = 0; k0 < K; k0 += 32) {
    V16H a, b0, b1;
    const int ka = k0 + half * 16;              // contiguous k = 16*half + s (matches B)
    a.p[0] = *(const v8h*)(aptr + ka);
    a.p[1] = *(const v8h*)(aptr + ka + 8);
    const int kb = k0 + half * 16;              // contiguous k = 16*half + s (matches A)
    b0.p[0] = *(const v8h*)(bptr0 + kb);
    b0.p[1] = *(const v8h*)(bptr0 + kb + 8);
    b1.p[0] = *(const v8h*)(bptr1 + kb);
    b1.p[1] = *(const v8h*)(bptr1 + kb + 8);
    c0 = __builtin_amdgcn_wmma_f32_16x16x32_f16(false, a.v, false, b0.v,
                                                (short)0, c0, false, false);
    c1 = __builtin_amdgcn_wmma_f32_16x16x32_f16(false, a.v, false, b1.v,
                                                (short)0, c1, false, false);
  }
  acc0.v = c0; acc1.v = c1;
  // C/D layout: VGPR j -> M = tm*16 + half*8 + j, N = tile_n_base + (lane&15)
#pragma unroll
  for (int j = 0; j < 8; ++j) {
    int m = tm * 16 + half * 8 + j;
    if (m >= M) continue;
#pragma unroll
    for (int t = 0; t < 2; ++t) {
      int n = tn * 32 + t * 16 + l15;
      float v = t ? acc1.f[j] : acc0.f[j];
      if (bias) v += bias[n];
      if (act == 1) v = v > 0.0f ? v : 0.0f;
      else if (act == 2) v = v > 0.0f ? v : SLOPE * v;
      C[(size_t)m * ldc + n] = v;
    }
  }
}

// ---------------- node memory update / snode ----------------
__global__ void k_recent(const float* __restrict__ memory, const float* __restrict__ last,
                         const float* __restrict__ t_u, const float* __restrict__ cnt,
                         const float* __restrict__ msg, const float* __restrict__ now_p,
                         float* __restrict__ recent) {
  size_t i = blockIdx.x * (size_t)blockDim.x + threadIdx.x;
  if (i >= (size_t)cN * 512) return;
  int n = (int)(i / 512), l = (int)((i % 512) / 128);
  float now = now_p[0];
  bool m = cnt[n] > 0.0f;
  float tu = t_u[n], lu = last[n];
  float um = m ? memory[i] * __expf(-c_lambs[l] * (tu - lu)) + msg[i] : memory[i];
  float ul = m ? tu : lu;
  recent[i] = um * __expf(-c_lambs[l] * (now - ul));
}

__global__ void k_snode(const float* __restrict__ memory, const float* __restrict__ last,
                        const float* __restrict__ begin_p, float* __restrict__ snode) {
  size_t i = blockIdx.x * (size_t)blockDim.x + threadIdx.x;
  if (i >= (size_t)cN * 512) return;
  int n = (int)(i / 512), l = (int)((i % 512) / 128);
  snode[i] = memory[i] * __expf(-c_lambs[l] * (begin_p[0] - last[n]));
}

// ---------------- attention ----------------
// A_r[r][h][n] = dot(qr[r][h*16..], kproj[n][h*16..]) / 4
__global__ void k_Ar(const float* __restrict__ qr, const float* __restrict__ kproj,
                     float* __restrict__ A_r) {
  int idx = blockIdx.x * blockDim.x + threadIdx.x;
  if (idx >= cH * cN) return;
  int h = idx / cN, n = idx % cN;
  float kv[16];
#pragma unroll
  for (int f = 0; f < 16; ++f) kv[f] = kproj[(size_t)n * 128 + h * 16 + f];
  for (int r = 0; r < cR; ++r) {
    float a = 0.0f;
#pragma unroll
    for (int f = 0; f < 16; ++f) a += qr[r * 128 + h * 16 + f] * kv[f];
    A_r[((size_t)r * cH + h) * cN + n] = a * 0.25f;
  }
}

__device__ inline float blockRed(float v, float* sh, int isMax) {
#pragma unroll
  for (int o = 16; o > 0; o >>= 1) {
    float t = __shfl_xor(v, o, 32);
    v = isMax ? fmaxf(v, t) : (v + t);
  }
  int wid = threadIdx.x >> 5, lid = threadIdx.x & 31, nw = blockDim.x >> 5;
  if (lid == 0) sh[wid] = v;
  __syncthreads();
  if (threadIdx.x == 0) {
    float r = sh[0];
    for (int w = 1; w < nw; ++w) r = isMax ? fmaxf(r, sh[w]) : (r + sh[w]);
    sh[0] = r;
  }
  __syncthreads();
  float r = sh[0];
  __syncthreads();
  return r;
}

// block per (r,h): masked softmax over n (axis=2) + rm_mid accumulation
__global__ __launch_bounds__(256)
void k_attn_rm(const float* __restrict__ A_r, const float* __restrict__ cnt,
               const float* __restrict__ v, float* __restrict__ rm_mid) {
  __shared__ float sh[8];
  int r = blockIdx.x, h = blockIdx.y;
  const float* arow = A_r + ((size_t)r * cH + h) * cN;
  float mx = NEGV;
  for (int n = threadIdx.x; n < cN; n += 256)
    mx = fmaxf(mx, cnt[n] > 0.0f ? arow[n] : NEGV);
  float smax = blockRed(mx, sh, 1);
  float sm = 0.0f;
  for (int n = threadIdx.x; n < cN; n += 256)
    sm += __expf((cnt[n] > 0.0f ? arow[n] : NEGV) - smax);
  float inv = 1.0f / blockRed(sm, sh, 0);
  float acc[64];
#pragma unroll
  for (int i = 0; i < 64; ++i) acc[i] = 0.0f;
  for (int n = threadIdx.x; n < cN; n += 256) {
    float p = __expf((cnt[n] > 0.0f ? arow[n] : NEGV) - smax) * inv;
    const float* vrow = v + (size_t)n * 512 + h * 16;   // v[n][l*128 + h*16 + f]
#pragma unroll
    for (int l = 0; l < 4; ++l)
#pragma unroll
      for (int f = 0; f < 16; ++f) acc[l * 16 + f] += p * vrow[l * 128 + f];
  }
#pragma unroll
  for (int i = 0; i < 64; ++i) {
    float s = blockRed(acc[i], sh, 0);
    if (threadIdx.x == 0)
      rm_mid[((size_t)r * 4 + (i >> 4)) * 128 + h * 16 + (i & 15)] = s;
  }
}

// r2n[r][n] = mean_h softmax_r(A_r[:,h,n]); gsum[n] = sum_r r2n[r][n]
__global__ void k_r2n(const float* __restrict__ A_r, float* __restrict__ r2n,
                      float* __restrict__ gsum) {
  int n = blockIdx.x * blockDim.x + threadIdx.x;
  if (n >= cN) return;
  float mx[8], sm[8];
#pragma unroll
  for (int h = 0; h < 8; ++h) { mx[h] = NEGV; sm[h] = 0.0f; }
  for (int r = 0; r < cR; ++r)
#pragma unroll
    for (int h = 0; h < 8; ++h)
      mx[h] = fmaxf(mx[h], A_r[((size_t)r * cH + h) * cN + n]);
  for (int r = 0; r < cR; ++r)
#pragma unroll
    for (int h = 0; h < 8; ++h)
      sm[h] += __expf(A_r[((size_t)r * cH + h) * cN + n] - mx[h]);
  float g = 0.0f;
  for (int r = 0; r < cR; ++r) {
    float a = 0.0f;
#pragma unroll
    for (int h = 0; h < 8; ++h)
      a += __expf(A_r[((size_t)r * cH + h) * cN + n] - mx[h]) / sm[h];
    a *= 0.125f;
    r2n[(size_t)r * cN + n] = a;
    g += a;
  }
  gsum[n] = g;
}

// A_g + softmax over r + gm_mid; thread per (l, h*16+f), 512 threads total
__global__ void k_graph_attn(const float* __restrict__ qg, const float* __restrict__ kg,
                             const float* __restrict__ vg, float* __restrict__ gm_mid) {
  int idx = blockIdx.x * blockDim.x + threadIdx.x;
  if (idx >= 4 * 128) return;
  int l = idx >> 7, c = idx & 127, h = c >> 4;
  float mx = NEGV;
  for (int r = 0; r < cR; ++r) {
    float s = 0.0f;
#pragma unroll
    for (int f = 0; f < 16; ++f) s += qg[h * 16 + f] * kg[r * 128 + h * 16 + f];
    mx = fmaxf(mx, s * 0.25f);
  }
  float sm = 0.0f;
  for (int r = 0; r < cR; ++r) {
    float s = 0.0f;
#pragma unroll
    for (int f = 0; f < 16; ++f) s += qg[h * 16 + f] * kg[r * 128 + h * 16 + f];
    sm += __expf(s * 0.25f - mx);
  }
  float acc = 0.0f;
  for (int r = 0; r < cR; ++r) {
    float s = 0.0f;
#pragma unroll
    for (int f = 0; f < 16; ++f) s += qg[h * 16 + f] * kg[r * 128 + h * 16 + f];
    acc += (__expf(s * 0.25f - mx) / sm) * vg[((size_t)r * 4 + l) * 128 + c];
  }
  gm_mid[idx] = acc;
}

// region / graph recent memories
__global__ void k_recent_rg(const float* __restrict__ rmem, const float* __restrict__ rlast,
                            const float* __restrict__ gmem, const float* __restrict__ glast,
                            const float* __restrict__ rmsg, const float* __restrict__ gmsg,
                            const float* __restrict__ now_p,
                            float* __restrict__ recent_r, float* __restrict__ recent_g) {
  int i = blockIdx.x * blockDim.x + threadIdx.x;
  float now = now_p[0];
  if (i < cR * 512) {
    int r = i / 512, l = (i % 512) / 128;
    recent_r[i] = rmem[i] * __expf(-c_lambs[l] * (now - rlast[r])) + rmsg[i];
  } else if (i < cR * 512 + 512) {
    int j = i - cR * 512, l = j / 128;
    recent_g[j] = gmem[j] * __expf(-c_lambs[l] * (now - glast[0])) + gmsg[j];
  }
}

// graph_node[n][c] = gsum[n] * recent_g[c]   (exact: softmax over singleton axis == 1)
__global__ void k_graph_node(const float* __restrict__ gsum, const float* __restrict__ rg,
                             float* __restrict__ gn) {
  size_t i = blockIdx.x * (size_t)blockDim.x + threadIdx.x;
  if (i >= (size_t)cN * 512) return;
  gn[i] = gsum[i / 512] * rg[i % 512];
}

__global__ void k_final(const float* __restrict__ stat, const float* __restrict__ spat,
                        const float* __restrict__ lam_p, float* __restrict__ out) {
  size_t i = blockIdx.x * (size_t)blockDim.x + threadIdx.x;
  if (i >= (size_t)cN * cD) return;
  float lam = lam_p[0];
  out[i] = lam * stat[i] + (1.0f - lam) * spat[i];
}

// ---------------- host launcher ----------------
extern "C" void kernel_launch(void* const* d_in, const int* in_sizes, int n_in,
                              void* d_out, int out_size, void* d_ws, size_t ws_size,
                              hipStream_t stream) {
  (void)in_sizes; (void)n_in; (void)out_size; (void)ws_size;
  const int*   src   = (const int*)d_in[0];
  const int*   tgt   = (const int*)d_in[1];
  const float* etd   = (const float*)d_in[2];
  const float* ts    = (const float*)d_in[3];
  const float* nowp  = (const float*)d_in[4];
  const float* begp  = (const float*)d_in[5];
  const float* mem   = (const float*)d_in[7];
  const float* last  = (const float*)d_in[8];
  const float* rmem  = (const float*)d_in[9];
  const float* gmem  = (const float*)d_in[10];
  const float* rlast = (const float*)d_in[11];
  const float* glast = (const float*)d_in[12];
  const float* feat  = (const float*)d_in[13];
  const float* statE = (const float*)d_in[14];
  const float* Wm1 = (const float*)d_in[15]; const float* bm1 = (const float*)d_in[16];
  const float* Wm2 = (const float*)d_in[17]; const float* bm2 = (const float*)d_in[18];
  const float* Qr  = (const float*)d_in[19]; const float* Kr  = (const float*)d_in[20];
  const float* Vr  = (const float*)d_in[21]; const float* Qg  = (const float*)d_in[22];
  const float* Kg  = (const float*)d_in[23]; const float* Vg  = (const float*)d_in[24];
  const float* Wr1 = (const float*)d_in[25]; const float* br1 = (const float*)d_in[26];
  const float* Wr2 = (const float*)d_in[27]; const float* br2 = (const float*)d_in[28];
  const float* Wg1 = (const float*)d_in[29]; const float* bg1 = (const float*)d_in[30];
  const float* Wg2 = (const float*)d_in[31]; const float* bg2 = (const float*)d_in[32];
  const float* We  = (const float*)d_in[33]; const float* be  = (const float*)d_in[34];
  const float* Ws  = (const float*)d_in[35]; const float* bs  = (const float*)d_in[36];
  const float* lam = (const float*)d_in[37];
  float* out = (float*)d_out;

  // ---- workspace carve ----
  char* p = (char*)d_ws;
  auto alloc = [&](size_t bytes) -> void* {
    void* r = (void*)p; p += (bytes + 255) & ~(size_t)255; return r;
  };
  const size_t NL = (size_t)cN * cL;          // 40000
  const size_t NA = NL * RAW1;                // agg elements
  float* t_u  = (float*)alloc((2 * (size_t)cN + NA) * 4);   // t_u | cnt | agg contiguous
  float* cnt  = t_u + cN;
  float* agg  = cnt + cN;
  float* wdec = (float*)alloc((size_t)cE * 4 * 4);
  float* wts  = (float*)alloc((size_t)cE * 4 * 4);
  // hidden + hiddenB overlay the agg region once agg is staged to f16
  float*    hidden  = agg;                                   // 40000x320 f32
  _Float16* hiddenB = (_Float16*)(agg + NL * 320);           // 40000x320 f16
  _Float16* normA = (_Float16*)alloc(NL * RAW * 2);          // 40000x640
  _Float16* aggmA = (_Float16*)alloc(NL * RAW * 2);          // 40000x640
  float* msg    = (float*)alloc(NL * 128 * 4);
  float* recent = (float*)alloc((size_t)cN * 512 * 4);
  float* snode  = (float*)alloc((size_t)cN * 512 * 4);
  _Float16* snodeB = (_Float16*)alloc((size_t)cN * 512 * 2);
  float* kproj = (float*)alloc((size_t)cN * 128 * 4);
  float* qr = (float*)alloc(cR * 128 * 4);
  float* kg = (float*)alloc(cR * 128 * 4);
  float* qg = (float*)alloc(16 * 128 * 4);                   // M=1 GEMM tile writes row 0
  float* A_r = (float*)alloc((size_t)cR * cH * cN * 4);
  float* vmat = (float*)alloc(NL * 128 * 4);
  float* rm_mid = (float*)alloc(400 * 128 * 4);
  _Float16* rm_midB = (_Float16*)alloc(400 * 128 * 2);
  float* reg_h = (float*)alloc(400 * 128 * 4);
  _Float16* reg_hB = (_Float16*)alloc(400 * 128 * 2);
  float* rmsg = (float*)alloc(400 * 128 * 4);
  float* vg_out = (float*)alloc(400 * 128 * 4);
  float* gm_mid = (float*)alloc(16 * 128 * 4);
  _Float16* gm_midB = (_Float16*)alloc(16 * 128 * 2);
  float* g_h = (float*)alloc(16 * 128 * 4);
  _Float16* g_hB = (_Float16*)alloc(16 * 128 * 2);
  float* gmsg = (float*)alloc(16 * 128 * 4);
  float* recent_r = (float*)alloc(cR * 512 * 4);
  float* recent_g = (float*)alloc(512 * 4);
  float* r2n  = (float*)alloc((size_t)cR * cN * 4);
  float* gsum = (float*)alloc(cN * 4);
  float* region_node = (float*)alloc((size_t)cN * 512 * 4);
  float* graph_node  = (float*)alloc((size_t)cN * 512 * 4);
  float* et2 = (float*)alloc((size_t)cN * 384 * 4);
  float* spatial = (float*)alloc((size_t)cN * 128 * 4);
  _Float16* r2nTp = (_Float16*)alloc((size_t)cN * 128 * 2);  // r2n.T zero-padded K=100->128
  _Float16* rrTp  = (_Float16*)alloc(512 * 128 * 2);
  _Float16* rembB = (_Float16*)alloc(cR * 512 * 2);
  _Float16* gembB = (_Float16*)alloc(512 * 2);
  _Float16* Wm1T = (_Float16*)alloc(320 * 640 * 2);
  _Float16* Wm2T = (_Float16*)alloc(128 * 320 * 2);
  _Float16* QrT  = (_Float16*)alloc(128 * 512 * 2);
  _Float16* KrT  = (_Float16*)alloc(128 * 512 * 2);
  _Float16* VrT  = (_Float16*)alloc(128 * 640 * 2);
  _Float16* QgT  = (_Float16*)alloc(128 * 512 * 2);
  _Float16* KgT  = (_Float16*)alloc(128 * 512 * 2);
  _Float16* VgT  = (_Float16*)alloc(128 * 128 * 2);
  _Float16* Wr1T = (_Float16*)alloc(128 * 128 * 2);
  _Float16* Wr2T = (_Float16*)alloc(128 * 128 * 2);
  _Float16* Wg1T = (_Float16*)alloc(128 * 128 * 2);
  _Float16* Wg2T = (_Float16*)alloc(128 * 128 * 2);
  _Float16* WeT  = (_Float16*)alloc(128 * 512 * 2);
  _Float16* WsT  = (_Float16*)alloc(128 * 384 * 2);
  // late f16 stagings overlay the freed aggmA region (25.6M f16 >= 19.2M needed)
  _Float16* recentB = aggmA;
  _Float16* rnodeB  = aggmA + (size_t)cN * 512;
  _Float16* gnodeB  = aggmA + (size_t)2 * cN * 512;
  _Float16* et2B    = aggmA + (size_t)3 * cN * 512;

  auto gemm = [&](const _Float16* A, const _Float16* BT, const float* bias, float* C,
                  int M, int K, int Nc, int ldc, int act) {
    int tiles = ((M + 15) / 16) * (Nc / 32);
    k_wmma_gemm<<<dim3(divup(tiles, 4)), dim3(32, 4), 0, stream>>>(A, BT, bias, C,
                                                                   M, K, Nc, ldc, act);
  };
  auto cvt = [&](const float* s, _Float16* d, size_t n) {
    k_cvt_f16<<<divup((long)n, 256), 256, 0, stream>>>(s, d, n);
  };
  auto cvtT = [&](const float* s, _Float16* d, int K, int Nc, int Kpad) {
    k_cvt_T_pad<<<divup((long)Nc * Kpad, 256), 256, 0, stream>>>(s, d, K, Nc, Kpad);
  };

  // ---- weight staging (transpose + f16) ----
  cvtT(Wm1, Wm1T, 640, 320, 640);  cvtT(Wm2, Wm2T, 320, 128, 320);
  cvtT(Qr, QrT, 512, 128, 512);    cvtT(Kr, KrT, 512, 128, 512);
  cvtT(Vr, VrT, 640, 128, 640);    cvtT(Qg, QgT, 512, 128, 512);
  cvtT(Kg, KgT, 512, 128, 512);    cvtT(Vg, VgT, 128, 128, 128);
  cvtT(Wr1, Wr1T, 128, 128, 128);  cvtT(Wr2, Wr2T, 128, 128, 128);
  cvtT(Wg1, Wg1T, 128, 128, 128);  cvtT(Wg2, Wg2T, 128, 128, 128);
  cvtT(We, WeT, 512, 128, 512);    cvtT(Ws, WsT, 384, 128, 384);
  cvt(rmem, rembB, (size_t)cR * 512);
  cvt(gmem, gembB, 512);

  // ---- stage 1: segment reductions + fused scatter ----
  size_t zn = 2 * (size_t)cN + NA;
  k_zero<<<divup((long)zn, 256), 256, 0, stream>>>(t_u, zn);
  k_seg<<<divup(cE, 256), 256, 0, stream>>>(ts, src, t_u, cnt, cE);
  k_weights<<<divup(cE, 256), 256, 0, stream>>>(ts, etd, src, t_u, wdec, wts, cE);
  k_scatter<<<dim3(cE, 6), 128, 0, stream>>>(mem, feat, src, tgt, wdec, wts, agg);

  // ---- stage agg to f16 (normalized + raw), freeing agg region ----
  k_aggm_f16<<<divup((long)NL * RAW, 256), 256, 0, stream>>>(agg, normA, (int)NL, 1);
  k_aggm_f16<<<divup((long)NL * RAW, 256), 256, 0, stream>>>(agg, aggmA, (int)NL, 0);

  // ---- message MLP (WMMA) ----
  gemm(aggmA, Wm1T, bm1, hidden, (int)NL, 640, 320, 320, 1);
  cvt(hidden, hiddenB, NL * 320);
  gemm(hiddenB, Wm2T, bm2, msg, (int)NL, 320, 128, 128, 0);
  gemm(normA, VrT, nullptr, vmat, (int)NL, 640, 128, 128, 0);   // v = norm @ Vr

  // ---- node recent + snode ----
  k_recent<<<divup((long)cN * 512, 256), 256, 0, stream>>>(mem, last, t_u, cnt, msg,
                                                           nowp, recent);
  k_snode<<<divup((long)cN * 512, 256), 256, 0, stream>>>(mem, last, begp, snode);
  cvt(snode, snodeB, (size_t)cN * 512);

  // ---- projections + attention ----
  gemm(snodeB, KrT, nullptr, kproj, cN, 512, 128, 128, 0);
  gemm(rembB, QrT, nullptr, qr, cR, 512, 128, 128, 0);
  gemm(rembB, KgT, nullptr, kg, cR, 512, 128, 128, 0);
  gemm(gembB, QgT, nullptr, qg, 1, 512, 128, 128, 0);
  k_Ar<<<divup(cH * cN, 256), 256, 0, stream>>>(qr, kproj, A_r);
  k_attn_rm<<<dim3(cR, cH), 256, 0, stream>>>(A_r, cnt, vmat, rm_mid);
  k_r2n<<<divup(cN, 256), 256, 0, stream>>>(A_r, r2n, gsum);

  // ---- region / graph messages ----
  cvt(rm_mid, rm_midB, 400 * 128);
  gemm(rm_midB, Wr1T, br1, reg_h, 400, 128, 128, 128, 2);
  cvt(reg_h, reg_hB, 400 * 128);
  gemm(reg_hB, Wr2T, br2, rmsg, 400, 128, 128, 128, 2);
  gemm(rm_midB, VgT, nullptr, vg_out, 400, 128, 128, 128, 0);
  k_graph_attn<<<2, 256, 0, stream>>>(qg, kg, vg_out, gm_mid);
  cvt(gm_mid, gm_midB, 4 * 128);
  gemm(gm_midB, Wg1T, bg1, g_h, 4, 128, 128, 128, 2);
  cvt(g_h, g_hB, 4 * 128);
  gemm(g_hB, Wg2T, bg2, gmsg, 4, 128, 128, 128, 2);

  // ---- recent region/graph + node-level mixes ----
  k_recent_rg<<<divup(cR * 512 + 512, 256), 256, 0, stream>>>(rmem, rlast, gmem, glast,
                                                              rmsg, gmsg, nowp,
                                                              recent_r, recent_g);
  cvtT(r2n, r2nTp, cR, cN, 128);          // (N x 128) = r2n.T zero-padded
  cvtT(recent_r, rrTp, cR, 512, 128);     // (512 x 128) = recent_r.T zero-padded
  gemm(r2nTp, rrTp, nullptr, region_node, cN, 128, 512, 512, 0);
  k_graph_node<<<divup((long)cN * 512, 256), 256, 0, stream>>>(gsum, recent_g, graph_node);

  // ---- embedding head ----
  cvt(recent, recentB, (size_t)cN * 512);
  cvt(region_node, rnodeB, (size_t)cN * 512);
  cvt(graph_node, gnodeB, (size_t)cN * 512);
  gemm(recentB, WeT, be, et2 + 0,   cN, 512, 128, 384, 2);
  gemm(rnodeB,  WeT, be, et2 + 128, cN, 512, 128, 384, 2);
  gemm(gnodeB,  WeT, be, et2 + 256, cN, 512, 128, 384, 2);
  cvt(et2, et2B, (size_t)cN * 384);
  gemm(et2B, WsT, bs, spatial, cN, 384, 128, 128, 2);
  k_final<<<divup((long)cN * cD, 256), 256, 0, stream>>>(statE, spatial, lam, out);
}
